// VectorQuantizerEMA_7533372637763
// MI455X (gfx1250) — matrix operand order repaired and checked by reading, not compiled, not run on hardware
//
#include <hip/hip_runtime.h>
#include <hip/hip_bf16.h>

typedef __bf16 bf16_t;
typedef __attribute__((ext_vector_type(16))) __bf16 v16bf;
typedef __attribute__((ext_vector_type(8)))  __bf16 v8bf;
typedef __attribute__((ext_vector_type(8)))  float  v8f;
typedef __attribute__((ext_vector_type(4)))  int    v4i;

#define D_DIM      512
#define K_CODES    8192
#define N_ROWS     16384
#define COLS_BLK   64
#define LDS_STRIDE 520   // 512 bf16 + 8 pad -> 1040B row stride, kills LDS bank conflicts

// --------------------------------------------------------------------------
// gfx1250 async memory->LDS path (ASYNCcnt-tracked, no VGPR round trip)
// --------------------------------------------------------------------------
#if defined(__AMDGCN__)
#if __has_builtin(__builtin_amdgcn_global_load_async_to_lds_b128)
#define HAVE_ASYNC_LDS 1
#else
#define HAVE_ASYNC_LDS 0
#pragma message("gfx1250 async-to-LDS builtin NOT available; falling back to VGPR staging")
#endif
#else
#define HAVE_ASYNC_LDS 0
#endif

#if HAVE_ASYNC_LDS
typedef __attribute__((address_space(1))) v4i* gptr_v4i;
typedef __attribute__((address_space(3))) v4i* lptr_v4i;
#endif

__device__ __forceinline__ void async_copy_b128(const void* g, void* l) {
#if HAVE_ASYNC_LDS
    __builtin_amdgcn_global_load_async_to_lds_b128(
        (gptr_v4i)(void*)g, (lptr_v4i)l, 0, 0);
#else
    *(uint4*)l = *(const uint4*)g;
#endif
}

__device__ __forceinline__ void wait_async0() {
#if defined(__AMDGCN__)
#if __has_builtin(__builtin_amdgcn_s_wait_asynccnt)
    __builtin_amdgcn_s_wait_asynccnt(0);
#elif HAVE_ASYNC_LDS
    asm volatile("s_wait_asynccnt 0x0" ::: "memory");
#endif
#endif
}

// ---------------------------------------------------------------------------
// ws layout (bytes):
//   [0,              8388608)   E in bf16            (8192*512*2)
//   [8388608,        8421376)   esq  : ||e_k||^2 f32 (8192*4)
//   [8421376,        8486912)   idx  : argmin  i32   (16384*4)
//   [8486912,        8519680)   counts f32           (8192*4)
//   [8519680,        8519684)   sumsq f32
// ---------------------------------------------------------------------------

// Kernel 0: convert codebook to bf16, compute ||e_k||^2, zero accumulators.
__global__ void vq_prep(const float* __restrict__ E, bf16_t* __restrict__ Ebf,
                        float* __restrict__ esq, float* __restrict__ counts,
                        float* __restrict__ sumsq) {
    const int k = blockIdx.x;          // one codebook row per block
    const int t = threadIdx.x;         // 256 threads
    const float* er = E + (size_t)k * D_DIM;
    float s = 0.f;
    for (int d = t; d < D_DIM; d += 256) {
        const float v = er[d];
        Ebf[(size_t)k * D_DIM + d] = (bf16_t)v;
        s += v * v;
    }
    __shared__ float red[8];
    #pragma unroll
    for (int off = 16; off >= 1; off >>= 1) s += __shfl_xor(s, off, 32);
    if ((t & 31) == 0) red[t >> 5] = s;
    __syncthreads();
    if (t == 0) {
        float tot = 0.f;
        #pragma unroll
        for (int w = 0; w < 8; ++w) tot += red[w];
        esq[k]    = tot;
        counts[k] = 0.f;
        if (k == 0) *sumsq = 0.f;
    }
}

// B-fragment fetch from LDS: two runs of 8 bf16 per lane, 16 apart (two b128 loads).
__device__ __forceinline__ v16bf loadB(const bf16_t* p, int kf) {
    const v8bf lo = *(const v8bf*)(p + kf * 32);
    const v8bf hi = *(const v8bf*)(p + kf * 32 + 16);
    v16bf b;
    #pragma unroll
    for (int e = 0; e < 8; ++e) { b[e] = lo[e]; b[e + 8] = hi[e]; }
    return b;
}

// Kernel 1: WMMA bf16 score GEMM + running argmin.
// Wave owns 16 z-rows; all A fragments live in VGPRs for the whole kernel.
// E codebook blocks are double-buffered in LDS via async-to-LDS copies.
__global__ void __launch_bounds__(256)
vq_argmin(const float* __restrict__ z, const bf16_t* __restrict__ Ebf,
          const float* __restrict__ esq, int* __restrict__ idxout) {
    __shared__ bf16_t ldsE[2][COLS_BLK * LDS_STRIDE];   // 2 x 66,560 B

    const int t    = threadIdx.x;
    const int lane = t & 31;
    const int wave = t >> 5;
    const int half = lane >> 4;     // 0: lanes 0-15, 1: lanes 16-31
    const int l15  = lane & 15;
    const int rowBase = blockIdx.x * 128 + wave * 16;
    const int myrow   = rowBase + l15;

    // ---- Load 16 A fragments (this wave's 16 rows x full D=512, bf16) -----
    // 16-bit A 16x32 layout: lane<16 holds K {kc..kc+7, kc+16..kc+23},
    //                        lane>=16 holds K {kc+8..kc+15, kc+24..kc+31}.
    v16bf a[16];
    const float* zr = z + (size_t)myrow * D_DIM;
    #pragma unroll
    for (int f = 0; f < 16; ++f) {
        const int kc = f * 32;
        const float4 p0 = *(const float4*)(zr + kc + half * 8);
        const float4 p1 = *(const float4*)(zr + kc + half * 8 + 4);
        const float4 p2 = *(const float4*)(zr + kc + 16 + half * 8);
        const float4 p3 = *(const float4*)(zr + kc + 16 + half * 8 + 4);
        a[f][0]  = (bf16_t)p0.x; a[f][1]  = (bf16_t)p0.y;
        a[f][2]  = (bf16_t)p0.z; a[f][3]  = (bf16_t)p0.w;
        a[f][4]  = (bf16_t)p1.x; a[f][5]  = (bf16_t)p1.y;
        a[f][6]  = (bf16_t)p1.z; a[f][7]  = (bf16_t)p1.w;
        a[f][8]  = (bf16_t)p2.x; a[f][9]  = (bf16_t)p2.y;
        a[f][10] = (bf16_t)p2.z; a[f][11] = (bf16_t)p2.w;
        a[f][12] = (bf16_t)p3.x; a[f][13] = (bf16_t)p3.y;
        a[f][14] = (bf16_t)p3.z; a[f][15] = (bf16_t)p3.w;
    }

    // Stage one 64-code block (64 rows x 1024B) into an LDS buffer, async.
    auto stage = [&](int cb, int buf) {
        const char* src = (const char*)Ebf + (size_t)cb * (D_DIM * 2);
        #pragma unroll
        for (int p = 0; p < 16; ++p) {
            const int i = t + p * 256;      // 4096 x 16B chunks
            const int r = i >> 6;           // 64 chunks per row
            const int c = i & 63;
            async_copy_b128(src + r * (D_DIM * 2) + c * 16,
                            (void*)&ldsE[buf][r * LDS_STRIDE + c * 8]);
        }
    };

    // Running per-lane argmin, matching the 16x16 f32 C layout:
    // acc VGPR j <-> row (j + half*8), column (colBase + nc*16 + l15).
    float minv[8];
    int   mini[8];
    #pragma unroll
    for (int j = 0; j < 8; ++j) { minv[j] = 3.4e38f; mini[j] = 0; }

    stage(0, 0);
    wait_async0();
    __syncthreads();

    for (int cb = 0; cb < K_CODES; cb += COLS_BLK) {
        const int buf = (cb >> 6) & 1;
        // Kick off the next block's copies; they overlap this block's WMMAs.
        if (cb + COLS_BLK < K_CODES) {
            stage(cb + COLS_BLK, buf ^ 1);
            __builtin_prefetch(esq + cb + COLS_BLK, 0, 0);
        }

        #pragma unroll
        for (int nc = 0; nc < 4; ++nc) {
            v8f acc = {};
            // 16-bit B 32x16 layout: lane<16: N=lane, K {kc..kc+7, kc+16..kc+23};
            //                        lane>=16: N=lane-16, K {kc+8..kc+15, kc+24..kc+31}.
            const bf16_t* brow = &ldsE[buf][(nc * 16 + l15) * LDS_STRIDE + half * 8];
            v16bf bcur = loadB(brow, 0);
            #pragma unroll
            for (int kf = 0; kf < 16; ++kf) {
                v16bf bnext = bcur;
                if (kf < 15) bnext = loadB(brow, kf + 1);   // prefetch next frag
                acc = __builtin_amdgcn_wmma_f32_16x16x32_bf16(
                        false, a[kf], false, bcur, (short)0, acc, false, false);
                bcur = bnext;
            }
            const int   n  = cb + nc * 16 + l15;
            const float eq = esq[n];
            #pragma unroll
            for (int j = 0; j < 8; ++j) {
                const float dist = eq - 2.0f * acc[j];   // ||e||^2 - 2 z.e
                if (dist < minv[j]) { minv[j] = dist; mini[j] = n; }
            }
        }
        wait_async0();      // next block's async copies have landed in LDS
        __syncthreads();    // all waves done reading buf / writing buf^1
    }

    // ---- Cross-lane argmin within each 16-lane half -----------------------
    #pragma unroll
    for (int j = 0; j < 8; ++j) {
        float v  = minv[j];
        int   ix = mini[j];
        #pragma unroll
        for (int off = 8; off >= 1; off >>= 1) {
            const float ov = __shfl_xor(v, off, 32);
            const int   oi = __shfl_xor(ix, off, 32);
            if (ov < v || (ov == v && oi < ix)) { v = ov; ix = oi; }
        }
        if (l15 == 0) idxout[rowBase + half * 8 + j] = ix;
    }
}

// Kernel 2: gather quantized rows, Sum (q-z)^2, histogram, emit indices.
__global__ void vq_gather(const float* __restrict__ z, const float* __restrict__ E,
                          const int* __restrict__ idx, float* __restrict__ out,
                          float* __restrict__ counts, float* __restrict__ sumsq) {
    const int r = blockIdx.x;
    const int t = threadIdx.x;
    const int code = idx[r];
    const float* er = E + (size_t)code * D_DIM;
    const float* zr = z + (size_t)r * D_DIM;
    float* qr = out + (size_t)r * D_DIM;
    float s = 0.f;
    for (int d = t; d < D_DIM; d += 256) {
        const float q  = er[d];
        const float zv = zr[d];
        qr[d] = q;                       // straight-through output == quantized
        const float df = q - zv;
        s += df * df;
    }
    __shared__ float red[8];
    #pragma unroll
    for (int off = 16; off >= 1; off >>= 1) s += __shfl_xor(s, off, 32);
    if ((t & 31) == 0) red[t >> 5] = s;
    __syncthreads();
    if (t == 0) {
        float tot = 0.f;
        #pragma unroll
        for (int w = 0; w < 8; ++w) tot += red[w];
        atomicAdd(sumsq, tot);
        atomicAdd(&counts[code], 1.0f);
        out[(size_t)N_ROWS * D_DIM + 2 + r] = (float)code;   // indices tail
    }
}

// Kernel 3: vq_loss and perplexity scalars.
__global__ void vq_finalize(const float* __restrict__ counts,
                            const float* __restrict__ sumsq,
                            float* __restrict__ out) {
    const int t = threadIdx.x;
    float h = 0.f;
    for (int k = t; k < K_CODES; k += 256) {
        const float p = counts[k] * (1.0f / (float)N_ROWS);
        h += p * __logf(p + 1e-10f);
    }
    __shared__ float red[8];
    #pragma unroll
    for (int off = 16; off >= 1; off >>= 1) h += __shfl_xor(h, off, 32);
    if ((t & 31) == 0) red[t >> 5] = h;
    __syncthreads();
    if (t == 0) {
        float tot = 0.f;
        #pragma unroll
        for (int w = 0; w < 8; ++w) tot += red[w];
        // vq_loss = (1 + 0.25) * mean((q - z)^2)
        out[(size_t)N_ROWS * D_DIM]     = 1.25f * (*sumsq) / (float)((size_t)N_ROWS * D_DIM);
        out[(size_t)N_ROWS * D_DIM + 1] = __expf(-tot);
    }
}

extern "C" void kernel_launch(void* const* d_in, const int* in_sizes, int n_in,
                              void* d_out, int out_size, void* d_ws, size_t ws_size,
                              hipStream_t stream) {
    const float* z = (const float*)d_in[0];   // [32,512,512] f32
    const float* E = (const float*)d_in[1];   // [8192,512]  f32
    float* out = (float*)d_out;
    char*  ws  = (char*)d_ws;

    bf16_t* Ebf    = (bf16_t*)ws;
    float*  esq    = (float*)(ws + 8388608);
    int*    idx    = (int*)  (ws + 8388608 + 32768);
    float*  counts = (float*)(ws + 8388608 + 32768 + 65536);
    float*  sumsq  = (float*)(ws + 8388608 + 32768 + 65536 + 32768);

    vq_prep    <<<K_CODES,      256, 0, stream>>>(E, Ebf, esq, counts, sumsq);
    vq_argmin  <<<N_ROWS / 128, 256, 0, stream>>>(z, Ebf, esq, idx);
    vq_gather  <<<N_ROWS,       256, 0, stream>>>(z, E, idx, out, counts, sumsq);
    vq_finalize<<<1,            256, 0, stream>>>(counts, sumsq, out);
}
